// Net_16063177687366
// MI455X (gfx1250) — compile-verified, matrix operand order checked by reading, not run on hardware
//
#include <hip/hip_runtime.h>
#include <hip/hip_bf16.h>

// ---------------- problem constants (match reference) ----------------
#define N_GENES 4264
#define IN_DIM  1613
#define N_RELS  4
#define DIM1    1340
#define DIM2    920
#define DIM3    740
#define N_DRUGS 4000
#define N_EDGES 100000
#define NN      (N_DRUGS + N_GENES)   // 8264

// ---------------- WMMA types ----------------
typedef __attribute__((ext_vector_type(16))) __bf16 v16bf;
typedef __attribute__((ext_vector_type(2)))  __bf16 v2bf;
typedef __attribute__((ext_vector_type(8)))  float  v8f;

// ---------------- GEMM tiling ----------------
#define BM   128
#define BN   128
#define BK   32
#define KPAD 48   // LDS row stride (elems): 96B -> every 16-elem fragment 32B aligned

static __device__ __forceinline__ int imin(int a, int b) { return a < b ? a : b; }

// C[M,N] = A[M,K] @ B[K,N]   (ACCUM: C += A@B, C preloaded into accumulators)
// A,B,C f32 row-major; A/B converted to bf16 on the fly; bf16 WMMA, f32 accum.
// Block 128x128, 8 wave32s: wave tile 32x64 = 2x4 WMMA 16x16x32.
template <bool ACCUM>
__global__ __launch_bounds__(256, 1)
void gemm_bf16_wmma(const float* __restrict__ A, const float* __restrict__ B,
                    float* __restrict__ C, int M, int N, int K)
{
    // double-buffered bf16 tiles
    __shared__ __bf16 As[2][BM][KPAD];
    __shared__ __bf16 Bs[2][BN][KPAD];   // transposed: Bs[b][n][k]

    const int tid  = threadIdx.x;
    const int lane = tid & 31;          // wave32
    const int wave = tid >> 5;          // 0..7
    const int wm   = wave >> 1;         // 0..3 -> M direction (32 rows each)
    const int wn   = wave & 1;          // 0..1 -> N direction (64 cols each)
    const int half = lane >> 4;         // K-half / M-half selector per ISA layout
    const int l16  = lane & 15;

    const int rowBase = blockIdx.y * BM;
    const int colBase = blockIdx.x * BN;

    // ---- accumulators (preload C when fusing "+ root") ----
    v8f acc[2][4];
#pragma unroll
    for (int i = 0; i < 2; ++i)
#pragma unroll
        for (int j = 0; j < 4; ++j) {
            if (ACCUM) {
                // clamped, branch-free: OOB lanes read duplicated valid data
                // that is never stored back.
                const int col = imin(colBase + wn * 64 + j * 16 + l16, N - 1);
#pragma unroll
                for (int v = 0; v < 8; ++v) {
                    const int row =
                        imin(rowBase + wm * 32 + i * 16 + half * 8 + v, M - 1);
                    acc[i][j][v] = C[(size_t)row * N + col];
                }
            } else {
                acc[i][j] = (v8f)0.0f;
            }
        }

    // ---- staging maps (pairs along K) ----
    // A tile: 128x32 = 2048 pairs, 8 per thread:
    //   row = (tid>>4) + i*16, colpair = (tid&15)*2
    const int aCol  = (tid & 15) * 2;
    const int aRow0 = tid >> 4;
    // B tile: 32x128 = 2048 pairs (pairs along k), 8 per thread:
    //   n = tid&127, k = (tid>>7)*2 + i*4
    const int bCol = tid & 127;
    const int bK0  = (tid >> 7) * 2;

    // advancing per-thread pointers (clamped at M/N edges: duplicated data
    // only feeds never-stored outputs). Bumped by constants each K-step so
    // the hot loop has zero 64-bit multiplies.
    const float* aP[8];
#pragma unroll
    for (int i = 0; i < 8; ++i)
        aP[i] = A + (size_t)imin(rowBase + aRow0 + i * 16, M - 1) * K + aCol;
    const float* bP[8];
    {
        const int bColC = imin(colBase + bCol, N - 1);
#pragma unroll
        for (int i = 0; i < 8; ++i)
            bP[i] = B + (size_t)(bK0 + i * 4) * N + bColC;
    }

    float av[8][2];
    float bv[8][2];

    // full-tile fetch: no compares, no clamps; immediate-offset second elems
    auto fetch_full = [&]() {
#pragma unroll
        for (int i = 0; i < 8; ++i) {
            av[i][0] = aP[i][0];
            av[i][1] = aP[i][1];
        }
#pragma unroll
        for (int i = 0; i < 8; ++i) {
            bv[i][0] = bP[i][0];
            bv[i][1] = bP[i][N];    // N*4 B fits 24-bit inst offset
        }
    };

    auto advance = [&]() {
#pragma unroll
        for (int i = 0; i < 8; ++i) aP[i] += BK;
#pragma unroll
        for (int i = 0; i < 8; ++i) bP[i] += (size_t)BK * N;
    };

    // ragged K tail off the already-advanced pointers: clamped relative
    // offsets, zero out-of-range K. Runs once.
    auto fetch_tail = [&](int k0) {
        const int gc = k0 + aCol;
        const int d0 = imin(gc, K - 1) - gc;             // 0 or negative
        const int d1 = imin(gc + 1, K - 1) - (gc + 1);
#pragma unroll
        for (int i = 0; i < 8; ++i) {
            const float v0 = aP[i][d0];
            const float v1 = aP[i][1 + d1];
            av[i][0] = (gc     < K) ? v0 : 0.0f;
            av[i][1] = (gc + 1 < K) ? v1 : 0.0f;
        }
#pragma unroll
        for (int i = 0; i < 8; ++i) {
            const int k  = k0 + bK0 + i * 4;
            const long e0 = (long)(imin(k,     K - 1) - k) * N;
            const long e1 = (long)(imin(k + 1, K - 1) - (k + 1)) * N;
            const float v0 = bP[i][e0];
            const float v1 = bP[i][N + e1];
            bv[i][0] = (k     < K) ? v0 : 0.0f;
            bv[i][1] = (k + 1 < K) ? v1 : 0.0f;
        }
    };

    // convert + packed ds_store_b32 into buffer `buf`
    auto stage = [&](int buf) {
#pragma unroll
        for (int i = 0; i < 8; ++i) {
            v2bf pk;
            pk[0] = (__bf16)av[i][0];
            pk[1] = (__bf16)av[i][1];
            *(v2bf*)&As[buf][aRow0 + i * 16][aCol] = pk;
        }
#pragma unroll
        for (int i = 0; i < 8; ++i) {
            v2bf pk;
            pk[0] = (__bf16)bv[i][0];
            pk[1] = (__bf16)bv[i][1];
            *(v2bf*)&Bs[buf][bCol][bK0 + i * 4] = pk;   // transposed store
        }
    };

    auto compute = [&](int buf) {
        v16bf afrag[2], bfrag[4];
#pragma unroll
        for (int i = 0; i < 2; ++i)
            afrag[i] =
                *(const v16bf*)&As[buf][wm * 32 + i * 16 + l16][half * 16];
#pragma unroll
        for (int j = 0; j < 4; ++j)
            bfrag[j] =
                *(const v16bf*)&Bs[buf][wn * 64 + j * 16 + l16][half * 16];
#pragma unroll
        for (int i = 0; i < 2; ++i)
#pragma unroll
            for (int j = 0; j < 4; ++j)
                acc[i][j] = __builtin_amdgcn_wmma_f32_16x16x32_bf16(
                    false, afrag[i], false, bfrag[j],
                    (short)0, acc[i][j], false, false);
    };

    const int Tfull   = K / BK;          // >= 1 for all our shapes (K >= 920)
    const bool ragged = (K % BK) != 0;

    // prologue: tile 0 is always full
    fetch_full();
    advance();
    stage(0);
    __syncthreads();

    int cur = 0;
    for (int t = 1; t < Tfull; ++t) {    // hot loop: pointer bumps only
        fetch_full();                    // loads issue early
        advance();
        compute(cur);                    // WMMAs hide load latency
        stage(cur ^ 1);                  // waits loads, cvt, ds_store_b32
        __syncthreads();                 // one barrier per K-step
        cur ^= 1;
    }
    if (ragged) {                        // single peeled masked iteration
        fetch_tail(Tfull * BK);
        compute(cur);
        stage(cur ^ 1);
        __syncthreads();
        cur ^= 1;
    }
    compute(cur);

    // ---- store D per ISA C/D layout (bounds-checked, runs once) ----
#pragma unroll
    for (int i = 0; i < 2; ++i)
#pragma unroll
        for (int j = 0; j < 4; ++j) {
            const int col = colBase + wn * 64 + j * 16 + l16;
#pragma unroll
            for (int v = 0; v < 8; ++v) {
                const int row = rowBase + wm * 32 + i * 16 + half * 8 + v;
                if (row < M && col < N)
                    C[(size_t)row * N + col] = acc[i][j][v];
            }
        }
}

// ---------------- helper kernels ----------------
__global__ void zero_f32(float* __restrict__ p, long n)
{
    long i = (long)blockIdx.x * blockDim.x + threadIdx.x;
    if (i < n) p[i] = 0.0f;
}

__global__ void zero_i32(int* __restrict__ p, long n)
{
    long i = (long)blockIdx.x * blockDim.x + threadIdx.x;
    if (i < n) p[i] = 0;
}

__global__ void concat_kernel(const float* __restrict__ x,
                              const float* __restrict__ gene,
                              float* __restrict__ xf)
{
    long i = (long)blockIdx.x * blockDim.x + threadIdx.x;
    const long nx = (long)N_DRUGS * IN_DIM;
    const long nt = (long)NN * IN_DIM;
    if (i < nt) xf[i] = (i < nx) ? x[i] : gene[i - nx];
}

__global__ void count_kernel(const int* __restrict__ ei,
                             const int* __restrict__ et,
                             int* __restrict__ cnt)
{
    int e = blockIdx.x * blockDim.x + threadIdx.x;
    if (e < N_EDGES) {
        int dst = ei[N_EDGES + e];
        atomicAdd(&cnt[dst * N_RELS + et[e]], 1);
    }
}

// acc[dst] += h_r[src] / cnt[dst, rel]   (mean + sum-over-relations fused)
__global__ void scatter_rel(const float* __restrict__ hr,
                            const int* __restrict__ ei,
                            const int* __restrict__ et,
                            const int* __restrict__ cnt,
                            float* __restrict__ acc, int D, int rel)
{
    const int e = blockIdx.x;
    if (et[e] != rel) return;
    const int src = ei[e];
    const int dst = ei[N_EDGES + e];
    int c = cnt[dst * N_RELS + rel];
    if (c < 1) c = 1;
    const float inv = 1.0f / (float)c;
    const float* __restrict__ s = hr + (size_t)src * D;
    float* __restrict__ d = acc + (size_t)dst * D;
    for (int i = threadIdx.x; i < D; i += blockDim.x)
        atomicAdd(&d[i], s[i] * inv);
}

__global__ void bias_relu(const float* __restrict__ in,
                          const float* __restrict__ bias,
                          float* __restrict__ out, long n, int D)
{
    long i = (long)blockIdx.x * blockDim.x + threadIdx.x;
    if (i < n) {
        const int d = (int)(i % D);
        const float v = in[i] + bias[d];
        out[i] = v > 0.0f ? v : 0.0f;
    }
}

// 740 -> 2 head + log_softmax, one wave32 per node
__global__ void head_kernel(const float* __restrict__ emb,
                            const float* __restrict__ w,   // (DIM3, 2) row-major
                            const float* __restrict__ b,
                            float* __restrict__ out)
{
    const int node = blockIdx.x * (blockDim.x >> 5) + (threadIdx.x >> 5);
    const int lane = threadIdx.x & 31;
    if (node >= NN) return;
    const float* __restrict__ e = emb + (size_t)node * DIM3;
    float z0 = 0.0f, z1 = 0.0f;
    for (int d = lane; d < DIM3; d += 32) {
        const float v = e[d];
        z0 += v * w[d * 2 + 0];
        z1 += v * w[d * 2 + 1];
    }
#pragma unroll
    for (int o = 16; o > 0; o >>= 1) {
        z0 += __shfl_xor(z0, o, 32);
        z1 += __shfl_xor(z1, o, 32);
    }
    if (lane == 0) {
        z0 += b[0]; z1 += b[1];
        const float m = fmaxf(z0, z1);
        const float lse = m + logf(expf(z0 - m) + expf(z1 - m));
        out[node * 2 + 0] = z0 - lse;
        out[node * 2 + 1] = z1 - lse;
    }
}

// ---------------- launch helpers ----------------
static inline void launch_gemm(bool accum, const float* A, const float* B, float* C,
                               int M, int N, int K, hipStream_t s)
{
    dim3 grid((N + BN - 1) / BN, (M + BM - 1) / BM);
    if (accum)
        gemm_bf16_wmma<true><<<grid, 256, 0, s>>>(A, B, C, M, N, K);
    else
        gemm_bf16_wmma<false><<<grid, 256, 0, s>>>(A, B, C, M, N, K);
}

static inline void launch_zero(float* p, long n, hipStream_t s)
{
    zero_f32<<<(int)((n + 255) / 256), 256, 0, s>>>(p, n);
}

extern "C" void kernel_launch(void* const* d_in, const int* in_sizes, int n_in,
                              void* d_out, int out_size, void* d_ws, size_t ws_size,
                              hipStream_t stream)
{
    (void)in_sizes; (void)n_in; (void)out_size; (void)ws_size;

    const float* x      = (const float*)d_in[0];
    const float* gene   = (const float*)d_in[1];
    const float* w_rel1 = (const float*)d_in[2];
    const float* root1  = (const float*)d_in[3];
    const float* b1     = (const float*)d_in[4];
    const float* w_rel2 = (const float*)d_in[5];
    const float* root2  = (const float*)d_in[6];
    const float* b2     = (const float*)d_in[7];
    const float* lin1_w = (const float*)d_in[8];
    const float* lin1_b = (const float*)d_in[9];
    const float* lin2_w = (const float*)d_in[10];
    const float* lin2_b = (const float*)d_in[11];
    const int*   ei     = (const int*)d_in[12];
    const int*   et     = (const int*)d_in[13];

    float* out_ls = (float*)d_out;                   // (NN, 2) log-softmax
    float* emb    = out_ls + (size_t)NN * 2;         // (NN, DIM3)

    // workspace layout
    char* w = (char*)d_ws;
    float* xf  = (float*)w;  w += sizeof(float) * (size_t)NN * IN_DIM;
    float* acc = (float*)w;  w += sizeof(float) * (size_t)NN * DIM1;
    float* hr  = (float*)w;  w += sizeof(float) * (size_t)NN * DIM1;
    float* h1  = (float*)w;  w += sizeof(float) * (size_t)NN * DIM1;
    int*   cnt = (int*)w;    w += sizeof(int)   * (size_t)NN * N_RELS;
    float* h2  = xf;  // xf dead after layer-1 root GEMM; reuse (NN*IN_DIM >= NN*DIM2)

    // 0) concat inputs
    {
        long n = (long)NN * IN_DIM;
        concat_kernel<<<(int)((n + 255) / 256), 256, 0, stream>>>(x, gene, xf);
    }

    // 1) per-(node, relation) edge counts (shared by both layers)
    zero_i32<<<(NN * N_RELS + 255) / 256, 256, 0, stream>>>(cnt, (long)NN * N_RELS);
    count_kernel<<<(N_EDGES + 255) / 256, 256, 0, stream>>>(ei, et, cnt);

    // 2) layer 1: xf(NN,1613) -> h1(NN,1340)
    launch_zero(acc, (long)NN * DIM1, stream);
    for (int r = 0; r < N_RELS; ++r) {
        launch_gemm(false, xf, w_rel1 + (size_t)r * IN_DIM * DIM1, hr,
                    NN, DIM1, IN_DIM, stream);
        scatter_rel<<<N_EDGES, 256, 0, stream>>>(hr, ei, et, cnt, acc, DIM1, r);
    }
    launch_gemm(true, xf, root1, acc, NN, DIM1, IN_DIM, stream);   // acc += xf@root1
    {
        long n = (long)NN * DIM1;
        bias_relu<<<(int)((n + 255) / 256), 256, 0, stream>>>(acc, b1, h1, n, DIM1);
    }

    // 3) layer 2: h1(NN,1340) -> h2(NN,920)
    launch_zero(acc, (long)NN * DIM2, stream);
    for (int r = 0; r < N_RELS; ++r) {
        launch_gemm(false, h1, w_rel2 + (size_t)r * DIM1 * DIM2, hr,
                    NN, DIM2, DIM1, stream);
        scatter_rel<<<N_EDGES, 256, 0, stream>>>(hr, ei, et, cnt, acc, DIM2, r);
    }
    launch_gemm(true, h1, root2, acc, NN, DIM2, DIM1, stream);     // acc += h1@root2
    {
        long n = (long)NN * DIM2;
        bias_relu<<<(int)((n + 255) / 256), 256, 0, stream>>>(acc, b2, h2, n, DIM2);
    }

    // 4) emb = relu(h2 @ lin1_w + lin1_b)  (written into d_out slice)
    launch_gemm(false, h2, lin1_w, hr, NN, DIM3, DIM2, stream);
    {
        long n = (long)NN * DIM3;
        bias_relu<<<(int)((n + 255) / 256), 256, 0, stream>>>(hr, lin1_b, emb, n, DIM3);
    }

    // 5) logits + log_softmax, one wave32 per node (8 nodes per 256-thread block)
    {
        const int nodes_per_block = 256 / 32;
        head_kernel<<<(NN + nodes_per_block - 1) / nodes_per_block, 256, 0, stream>>>(
            emb, lin2_w, lin2_b, out_ls);
    }
}
